// CN2LinkPredictor_51256139711065
// MI455X (gfx1250) — matrix operand order repaired and checked by reading, not compiled, not run on hardware
//
#include <hip/hip_runtime.h>
#include <hip/hip_bf16.h>
#include <stdint.h>

// Problem constants (match reference)
#define NNODES 8192
#define INCH   128
#define HIDDEN 256
#define NEDGE  4096
#define WORDS  256   // NNODES / 32

typedef __attribute__((ext_vector_type(16))) __bf16 v16bf;
typedef __attribute__((ext_vector_type(8)))  float  v8f;

static __device__ __forceinline__ unsigned short f2bf(float f) {
  unsigned u = __float_as_uint(f);
  unsigned r = u + 0x7FFFu + ((u >> 16) & 1u);   // round-to-nearest-even
  return (unsigned short)(r >> 16);
}

union BFrag { unsigned int ui[8]; v16bf v; };

// ---------------------------------------------------------------------------
// 1) Pack float adjacency (0.0/1.0) into bitset rows: bits[N][WORDS]
// ---------------------------------------------------------------------------
__global__ __launch_bounds__(256)
void k_pack(const float* __restrict__ adj, unsigned* __restrict__ bits) {
  long long g = (long long)blockIdx.x * blockDim.x + threadIdx.x;  // over N*N
  float v = adj[g];
  unsigned long long m = __ballot(v != 0.0f);     // wave32: low 32 bits
  if ((threadIdx.x & 31) == 0) bits[g >> 5] = (unsigned)m;
}

// ---------------------------------------------------------------------------
// 2) adj2 = (adj@adj > 0): row i = OR of adj rows of neighbors of i
//    One wave per row; lane l owns words {l + 32*t}.
// ---------------------------------------------------------------------------
__global__ __launch_bounds__(32)
void k_adj2(const unsigned* __restrict__ bits, unsigned* __restrict__ bits2) {
  int row  = blockIdx.x;
  int lane = threadIdx.x & 31;
  const unsigned* r = bits + (size_t)row * WORDS;
  unsigned my[8], acc[8];
#pragma unroll
  for (int t = 0; t < 8; ++t) { my[t] = r[lane + 32 * t]; acc[t] = 0u; }
  for (int t = 0; t < 8; ++t) {
    for (int sl = 0; sl < 32; ++sl) {
      unsigned w = __shfl(my[t], sl, 32);        // wave-uniform word
      int base = (sl + 32 * t) * 32;
      while (w) {
        int b = __ffs(w) - 1; w &= (w - 1);
        const unsigned* rk = bits + (size_t)(base + b) * WORDS;
#pragma unroll
        for (int q = 0; q < 8; ++q) acc[q] |= rk[lane + 32 * q];
      }
    }
  }
#pragma unroll
  for (int q = 0; q < 8; ++q) bits2[(size_t)row * WORDS + lane + 32 * q] = acc[q];
}

// ---------------------------------------------------------------------------
// 3) Weight prep: W[K][NOUT] f32  ->  Wt[NOUT][K] bf16 (transposed)
// ---------------------------------------------------------------------------
__global__ __launch_bounds__(256)
void k_prep_w(const float* __restrict__ W, unsigned short* __restrict__ Wt,
              int K, int NOUT) {
  int g = blockIdx.x * blockDim.x + threadIdx.x;
  if (g >= K * NOUT) return;
  int k = g / NOUT, n = g % NOUT;
  Wt[(size_t)n * K + k] = f2bf(W[g]);
}

// ---------------------------------------------------------------------------
// 4) Per-edge masked neighbor sums + pairwise product
//    One block (128 threads) per edge; thread t owns channel t.
// ---------------------------------------------------------------------------
__global__ __launch_bounds__(128)
void k_edge(const float* __restrict__ x,
            const unsigned* __restrict__ bits, const unsigned* __restrict__ bits2,
            const long long* __restrict__ tar,
            float* __restrict__ s1, float* __restrict__ s2,
            float* __restrict__ s3, float* __restrict__ s4,
            float* __restrict__ sij) {
  int e = blockIdx.x;
  int t = threadIdx.x;                       // channel 0..127
  long long i = tar[e], j = tar[NEDGE + e];
  __shared__ unsigned mAi[WORDS], mAj[WORDS], mBi[WORDS], mBj[WORDS];
  for (int w = t; w < WORDS; w += 128) {
    mAi[w] = bits [(size_t)i * WORDS + w];
    mAj[w] = bits [(size_t)j * WORDS + w];
    mBi[w] = bits2[(size_t)i * WORDS + w];
    mBj[w] = bits2[(size_t)j * WORDS + w];
  }
  __syncthreads();
  float a1 = 0.f, a2 = 0.f, a3 = 0.f, a4 = 0.f;
  for (int w = 0; w < WORDS; ++w) {
    unsigned ai = mAi[w], aj = mAj[w], bi = mBi[w], bj = mBj[w];
    unsigned m1 = ai & aj, m2 = ai & bj, m3 = bi & aj, m4 = bi & bj;
    unsigned u = m1 | m2 | m3 | m4;          // block-uniform
    while (u) {
      int b = __ffs(u) - 1; u &= (u - 1);
      unsigned bm = 1u << b;
      float xv = x[(size_t)(w * 32 + b) * INCH + t];   // coalesced 512B
      if (m1 & bm) a1 += xv;
      if (m2 & bm) a2 += xv;
      if (m3 & bm) a3 += xv;
      if (m4 & bm) a4 += xv;
    }
  }
  size_t o = (size_t)e * INCH + t;
  s1[o] = a1; s2[o] = a2; s3[o] = a3; s4[o] = a4;
  sij[o] = x[(size_t)i * INCH + t] * x[(size_t)j * INCH + t];
}

// ---------------------------------------------------------------------------
// 5) bf16 WMMA GEMM: C[M,NOUT] = act(A[M,K] @ W + bias)
//    A: f32 (converted on LDS store). Wt: [NOUT][K] bf16.
//    Block = 128 threads (4 waves), tile 64x128, BK=32.
//    Each wave owns a 32x64 sub-tile = 2x4 WMMA accumulators -> 8 WMMA / K-step.
// ---------------------------------------------------------------------------
#define BM 64
#define BN 128
#define LDA 36
#define LDB 36
__global__ __launch_bounds__(128)
void k_gemm(const float* __restrict__ A, const unsigned short* __restrict__ Wt,
            const float* __restrict__ bias, float* __restrict__ C,
            int M, int K, int NOUT, int relu) {
  __shared__ unsigned short As[BM * LDA];       // 4.5 KB
  __shared__ unsigned short Bs[BN * LDB];       // 9.0 KB
  int tid  = threadIdx.x;
  int wv   = tid >> 5;
  int lane = tid & 31;
  int half = lane >> 4;
  int r    = lane & 15;
  int wrow = (wv >> 1) * 32;                    // wave row offset in tile
  int wcol = (wv & 1) * 64;                     // wave col offset in tile
  int rowB = blockIdx.x * BM;
  int colB = blockIdx.y * BN;

  // Init accumulators with bias (column-broadcast: every v of a C tile shares
  // one column = colB + wcol + ni*16 + r).
  v8f acc[2][4];
#pragma unroll
  for (int ni = 0; ni < 4; ++ni) {
    float bv = bias[colB + wcol + ni * 16 + r];
    v8f b8 = {bv, bv, bv, bv, bv, bv, bv, bv};
    acc[0][ni] = b8;
    acc[1][ni] = b8;
  }

  for (int k0 = 0; k0 < K; k0 += 32) {
    // A tile: 64x32 f32 -> bf16
    for (int idx = tid; idx < BM * 32; idx += 128) {
      int rr = idx >> 5, kk = idx & 31;
      As[rr * LDA + kk] = f2bf(A[(size_t)(rowB + rr) * K + k0 + kk]);
    }
    // B tile: 128 cols x 32 k (already bf16, [NOUT][K] layout)
    for (int idx = tid; idx < BN * 32; idx += 128) {
      int rr = idx >> 5, kk = idx & 31;
      Bs[rr * LDB + kk] = Wt[(size_t)(colB + rr) * K + k0 + kk];
    }
    __syncthreads();

    BFrag aF[2], bF[4];
#pragma unroll
    for (int mi = 0; mi < 2; ++mi) {
      int arow = wrow + mi * 16 + r;
#pragma unroll
      for (int v = 0; v < 8; ++v) {
        int kb = (v < 4) ? (half * 8 + v * 2) : (16 + half * 8 + (v - 4) * 2);
        aF[mi].ui[v] = *reinterpret_cast<const unsigned int*>(&As[arow * LDA + kb]);
      }
    }
#pragma unroll
    for (int ni = 0; ni < 4; ++ni) {
      int bcol = wcol + ni * 16 + r;
#pragma unroll
      for (int v = 0; v < 8; ++v) {
        int kb = (v < 4) ? (half * 8 + v * 2) : (16 + half * 8 + (v - 4) * 2);
        bF[ni].ui[v] = *reinterpret_cast<const unsigned int*>(&Bs[bcol * LDB + kb]);
      }
    }
#pragma unroll
    for (int mi = 0; mi < 2; ++mi)
#pragma unroll
      for (int ni = 0; ni < 4; ++ni)
        acc[mi][ni] = __builtin_amdgcn_wmma_f32_16x16x32_bf16(
            false, aF[mi].v, false, bF[ni].v, (short)0, acc[mi][ni], false, false);
    __syncthreads();
  }

#pragma unroll
  for (int mi = 0; mi < 2; ++mi)
#pragma unroll
    for (int ni = 0; ni < 4; ++ni) {
      int cg = colB + wcol + ni * 16 + r;
#pragma unroll
      for (int v = 0; v < 8; ++v) {
        int rg = rowB + wrow + mi * 16 + v + 8 * half;
        float val = acc[mi][ni][v];
        if (relu) val = fmaxf(val, 0.f);
        C[(size_t)rg * NOUT + cg] = val;
      }
    }
}

// ---------------------------------------------------------------------------
// 6) Combine: z = a0*c1 + a1*c2*c3 + a2*c4 + beta*xij ; a = cumprod(sigmoid(alpha))
// ---------------------------------------------------------------------------
__global__ __launch_bounds__(256)
void k_comb(const float* __restrict__ c1, const float* __restrict__ c2,
            const float* __restrict__ c3, const float* __restrict__ c4,
            const float* __restrict__ xij,
            const float* __restrict__ alpha, const float* __restrict__ beta,
            float* __restrict__ z) {
  int g = blockIdx.x * blockDim.x + threadIdx.x;   // grid sized exactly
  float a0 = 1.f / (1.f + __expf(-alpha[0]));
  float a1 = a0 * (1.f / (1.f + __expf(-alpha[1])));
  float a2 = a1 * (1.f / (1.f + __expf(-alpha[2])));
  float bt = beta[0];
  z[g] = a0 * c1[g] + a1 * c2[g] * c3[g] + a2 * c4[g] + bt * xij[g];
}

// ---------------------------------------------------------------------------
// 7) Final 256->1 layer: out[e] = dot(H[e,:], W) + b ; wave-per-edge reduction
// ---------------------------------------------------------------------------
__global__ __launch_bounds__(256)
void k_final(const float* __restrict__ H, const float* __restrict__ W,
             const float* __restrict__ b, float* __restrict__ out) {
  int e    = blockIdx.x * 8 + (threadIdx.x >> 5);
  int lane = threadIdx.x & 31;
  float s = 0.f;
#pragma unroll
  for (int t = 0; t < 8; ++t)
    s += H[(size_t)e * HIDDEN + lane + 32 * t] * W[lane + 32 * t];
#pragma unroll
  for (int off = 16; off > 0; off >>= 1) s += __shfl_down(s, off, 32);
  if (lane == 0) out[e] = s + b[0];
}

// ---------------------------------------------------------------------------
extern "C" void kernel_launch(void* const* d_in, const int* in_sizes, int n_in,
                              void* d_out, int out_size, void* d_ws, size_t ws_size,
                              hipStream_t stream) {
  (void)in_sizes; (void)n_in; (void)out_size; (void)ws_size;
  const float*     x     = (const float*)d_in[0];
  const float*     adj   = (const float*)d_in[1];
  const long long* tar   = (const long long*)d_in[2];
  const float*     alpha = (const float*)d_in[3];
  const float*     beta  = (const float*)d_in[4];
  // p_cn1: 5..10, p_cn2: 11..16, p_cn4: 17..22, p_ij: 23..26, p_out: 27..30

  char* ws = (char*)d_ws;
  size_t off = 0;
  auto alloc = [&](size_t bytes) -> void* {
    void* p = ws + off;
    off = (off + bytes + 255) & ~((size_t)255);
    return p;
  };

  unsigned* adjBits  = (unsigned*)alloc((size_t)NNODES * WORDS * 4);
  unsigned* adj2Bits = (unsigned*)alloc((size_t)NNODES * WORDS * 4);

  // bf16 transposed weights
  unsigned short* wt[10];
  wt[0] = (unsigned short*)alloc((size_t)INCH   * HIDDEN * 2);  // cn1 W1
  wt[1] = (unsigned short*)alloc((size_t)HIDDEN * HIDDEN * 2);  // cn1 W2
  wt[2] = (unsigned short*)alloc((size_t)HIDDEN * HIDDEN * 2);  // cn1 W3
  wt[3] = (unsigned short*)alloc((size_t)INCH   * HIDDEN * 2);  // cn2 W1
  wt[4] = (unsigned short*)alloc((size_t)HIDDEN * HIDDEN * 2);  // cn2 W2
  wt[5] = (unsigned short*)alloc((size_t)HIDDEN * HIDDEN * 2);  // cn2 W3
  wt[6] = (unsigned short*)alloc((size_t)INCH   * HIDDEN * 2);  // cn4 W1
  wt[7] = (unsigned short*)alloc((size_t)HIDDEN * HIDDEN * 2);  // cn4 W2
  wt[8] = (unsigned short*)alloc((size_t)HIDDEN * HIDDEN * 2);  // cn4 W3
  wt[9] = (unsigned short*)alloc((size_t)INCH   * HIDDEN * 2);  // ij  W1
  unsigned short* wt_ij2  = (unsigned short*)alloc((size_t)HIDDEN * HIDDEN * 2);
  unsigned short* wt_out1 = (unsigned short*)alloc((size_t)HIDDEN * HIDDEN * 2);

  float* s1  = (float*)alloc((size_t)NEDGE * INCH * 4);
  float* s2  = (float*)alloc((size_t)NEDGE * INCH * 4);
  float* s3  = (float*)alloc((size_t)NEDGE * INCH * 4);
  float* s4  = (float*)alloc((size_t)NEDGE * INCH * 4);
  float* sij = (float*)alloc((size_t)NEDGE * INCH * 4);
  float* h1  = (float*)alloc((size_t)NEDGE * HIDDEN * 4);
  float* h2  = (float*)alloc((size_t)NEDGE * HIDDEN * 4);
  float* xcn[4];
  for (int b = 0; b < 4; ++b) xcn[b] = (float*)alloc((size_t)NEDGE * HIDDEN * 4);
  float* xijo = (float*)alloc((size_t)NEDGE * HIDDEN * 4);
  float* z    = (float*)alloc((size_t)NEDGE * HIDDEN * 4);

  // --- 1) pack + 2) adj2 ---
  {
    long long total = (long long)NNODES * NNODES;
    k_pack<<<(unsigned)(total / 256), 256, 0, stream>>>(adj, adjBits);
    k_adj2<<<NNODES, 32, 0, stream>>>(adjBits, adj2Bits);
  }

  // --- 3) weight prep ---
  auto prep = [&](const void* W, unsigned short* Wt, int K, int NOUT) {
    int n = K * NOUT;
    k_prep_w<<<(n + 255) / 256, 256, 0, stream>>>((const float*)W, Wt, K, NOUT);
  };
  prep(d_in[5],  wt[0], INCH,   HIDDEN);
  prep(d_in[7],  wt[1], HIDDEN, HIDDEN);
  prep(d_in[9],  wt[2], HIDDEN, HIDDEN);
  prep(d_in[11], wt[3], INCH,   HIDDEN);
  prep(d_in[13], wt[4], HIDDEN, HIDDEN);
  prep(d_in[15], wt[5], HIDDEN, HIDDEN);
  prep(d_in[17], wt[6], INCH,   HIDDEN);
  prep(d_in[19], wt[7], HIDDEN, HIDDEN);
  prep(d_in[21], wt[8], HIDDEN, HIDDEN);
  prep(d_in[23], wt[9], INCH,   HIDDEN);
  prep(d_in[25], wt_ij2,  HIDDEN, HIDDEN);
  prep(d_in[27], wt_out1, HIDDEN, HIDDEN);

  // --- 4) per-edge gathers ---
  k_edge<<<NEDGE, 128, 0, stream>>>(x, adjBits, adj2Bits, tar, s1, s2, s3, s4, sij);

  // --- 5) MLP chains via WMMA GEMMs ---
  auto gemm = [&](const float* A, const unsigned short* Wt, const void* bias,
                  float* C, int M, int K, int NOUT, int relu) {
    dim3 g(M / BM, NOUT / BN);
    k_gemm<<<g, 128, 0, stream>>>(A, Wt, (const float*)bias, C, M, K, NOUT, relu);
  };
  const float* src[4] = {s1, s2, s3, s4};
  // branch b: cn1 uses weights 0..2 / bias 6,8,10 ; cn2&cn3 share 3..5 / 12,14,16 ; cn4 6..8 / 18,20,22
  const int wbase[4] = {0, 3, 3, 6};
  const int bbase[4] = {6, 12, 12, 18};
  for (int b = 0; b < 4; ++b) {
    gemm(src[b], wt[wbase[b] + 0], d_in[bbase[b] + 0], h1, NEDGE, INCH,   HIDDEN, 1);
    gemm(h1,     wt[wbase[b] + 1], d_in[bbase[b] + 2], h2, NEDGE, HIDDEN, HIDDEN, 1);
    gemm(h2,     wt[wbase[b] + 2], d_in[bbase[b] + 4], xcn[b], NEDGE, HIDDEN, HIDDEN, 0);
  }
  // xij = mlp2(x[i]*x[j])
  gemm(sij, wt[9],   d_in[24], h1,   NEDGE, INCH,   HIDDEN, 1);
  gemm(h1,  wt_ij2,  d_in[26], xijo, NEDGE, HIDDEN, HIDDEN, 0);

  // --- 6) combine ---
  k_comb<<<(NEDGE * HIDDEN) / 256, 256, 0, stream>>>(
      xcn[0], xcn[1], xcn[2], xcn[3], xijo, alpha, beta, z);

  // --- 7) output mlp2 ---
  gemm(z, wt_out1, d_in[28], h1, NEDGE, HIDDEN, HIDDEN, 1);
  k_final<<<NEDGE / 8, 256, 0, stream>>>(h1, (const float*)d_in[29],
                                         (const float*)d_in[30], (float*)d_out);
}
